// TGCNCell_56057913147770
// MI455X (gfx1250) — matrix operand order repaired and checked by reading, not compiled, not run on hardware
//
#include <hip/hip_runtime.h>

#define NN 100000
#define HID 64
#define NE 1600000

typedef _Float16 h16;
typedef __attribute__((ext_vector_type(16))) h16 v16h;
typedef __attribute__((ext_vector_type(8)))  h16 v8h;
typedef __attribute__((ext_vector_type(8)))  float v8f;

// ---------------------------------------------------------------- utilities

__global__ void k_zero(float* __restrict__ p, size_t n) {
    size_t i = (size_t)blockIdx.x * blockDim.x + threadIdx.x;
    size_t s = (size_t)gridDim.x * blockDim.x;
    for (; i < n; i += s) p[i] = 0.0f;
}

__global__ void k_cvt_h(const float* __restrict__ in, h16* __restrict__ out, int n) {
    int i = blockIdx.x * blockDim.x + threadIdx.x;
    if (i < n) out[i] = (h16)in[i];
}

// ---------------------------------------------------------------- graph norm

__global__ void k_degree(const long long* __restrict__ src, float* __restrict__ deg, int E) {
    int e = blockIdx.x * blockDim.x + threadIdx.x;
    if (e < E) atomicAdd(&deg[src[e]], 1.0f);
}

__global__ void k_dinv(float* __restrict__ deg, int n) {
    int i = blockIdx.x * blockDim.x + threadIdx.x;
    if (i < n) {
        float d = deg[i];
        deg[i] = (d > 0.0f) ? rsqrtf(fmaxf(d, 1.0f)) : 0.0f;
    }
}

__global__ void k_edge_norm(const long long* __restrict__ src, const long long* __restrict__ dst,
                            const float* __restrict__ dinv, float* __restrict__ norm, int E) {
    int e = blockIdx.x * blockDim.x + threadIdx.x;
    if (e < E) norm[e] = -dinv[src[e]] * dinv[dst[e]];
}

// ------------------------------------------------------- sparse propagation

__global__ void k_prop2(const long long* __restrict__ src, const long long* __restrict__ dst,
                        const float* __restrict__ norm, const float* __restrict__ t,
                        float* __restrict__ out, int E) {
    int e = blockIdx.x * blockDim.x + threadIdx.x;
    if (e >= E) return;
    long long s = src[e], d = dst[e];
    float w = norm[e];
    float2 tv = ((const float2*)t)[s];
    atomicAdd(&out[d * 2 + 0], w * tv.x);
    atomicAdd(&out[d * 2 + 1], w * tv.y);
}

__global__ void k_prop64(const long long* __restrict__ src, const long long* __restrict__ dst,
                         const float* __restrict__ norm, const float* __restrict__ t,
                         float* __restrict__ out, long total) {
    long i = (long)blockIdx.x * blockDim.x + threadIdx.x;
    if (i >= total) return;
    long e = i >> 6;
    int  f = (int)(i & 63);
    long long s = src[e], d = dst[e];
    atomicAdd(&out[d * 64 + f], norm[e] * t[s * 64 + f]);
}

// -------------------------------------------------- input ChebConv (in=2)

__global__ void k_input_conv(const float* __restrict__ x, const float* __restrict__ tx1,
                             const float* __restrict__ tx2, const float* __restrict__ W1,
                             const float* __restrict__ b1,
                             h16* __restrict__ combined, h16* __restrict__ combC, int N) {
    long i = (long)blockIdx.x * blockDim.x + threadIdx.x;
    if (i >= (long)N * 64) return;
    long n = i >> 6;
    int  j = (int)(i & 63);
    float w00 = W1[0 * 128 + 0 * 64 + j], w01 = W1[0 * 128 + 64 + j];
    float w10 = W1[1 * 128 + 0 * 64 + j], w11 = W1[1 * 128 + 64 + j];
    float w20 = W1[2 * 128 + 0 * 64 + j], w21 = W1[2 * 128 + 64 + j];
    float x0 = x[n * 2], x1 = x[n * 2 + 1];
    float a0 = tx1[n * 2], a1 = tx1[n * 2 + 1];
    float c0 = tx2[n * 2], c1 = tx2[n * 2 + 1];
    float v = b1[j] + x0 * (w00 - w20) + x1 * (w01 - w21)
                    + a0 * w10 + a1 * w11
                    + 2.0f * (c0 * w20 + c1 * w21);
    combined[n * 128 + j] = (h16)v;
    combC[n * 128 + j]    = (h16)v;
}

// ------------------------------------- hidden ChebConv folding / packing

__global__ void k_pack_Ah(const float* __restrict__ h, const float* __restrict__ th1,
                          const float* __restrict__ th2, h16* __restrict__ Ah, int N) {
    long i = (long)blockIdx.x * blockDim.x + threadIdx.x;
    if (i >= (long)N * 64) return;
    long n = i >> 6;
    int  j = (int)(i & 63);
    Ah[n * 192 + j]       = (h16)h[i];
    Ah[n * 192 + 64 + j]  = (h16)th1[i];
    Ah[n * 192 + 128 + j] = (h16)th2[i];
}

__global__ void k_prep_weff(const float* __restrict__ W2, h16* __restrict__ Weff) {
    int i = blockIdx.x * blockDim.x + threadIdx.x;  // 0..4095
    if (i >= 64 * 64) return;
    Weff[i]            = (h16)(W2[i] - W2[2 * 4096 + i]);
    Weff[4096 + i]     = (h16)(W2[4096 + i]);
    Weff[2 * 4096 + i] = (h16)(2.0f * W2[2 * 4096 + i]);
}

// ---------------------------------------------------------------- WMMA GEMM
// C[N,64] = epilogue(A[N,K]_f16 @ B[K,64]_f16 + bias)
// One wave computes a 16x64 output tile with 4 f32 accumulators.
//
// A frag (16-bit 16x32): lane m=l&15, hh=l>>4. Per-lane K offsets are one
// contiguous 16-half run: [hh*16 .. hh*16+15]
//   -> two 16B global loads per fragment.
// B frag (16-bit 32x16): lane n=l&15, hh=l>>4; slot s (0..15) holds k = hh*16+s.
// B is staged into LDS pre-swizzled so each lane's fragment is 32 contiguous
// bytes: sB[((kblk*4+g)*32+lane)*16 + (k&15)] = B[k][g*16 + (lane&15)]
//   -> one 32B LDS load (2x ds_load_b128) per fragment.
// C/D: vgpr r -> row M = r + 8*(l>>4), col = l&15

__device__ __forceinline__ v16h load_a_frag(const h16* __restrict__ A, int lda,
                                            long row0, int kb, int lane) {
    int m = lane & 15, hh = lane >> 4;
    const h16* rp = A + (row0 + m) * (long)lda + kb + hh * 16;
    v8h lo = *(const v8h*)rp;
    v8h hi = *(const v8h*)(rp + 8);
    v16h a;
#pragma unroll
    for (int i = 0; i < 8; ++i) {
        a[i]     = lo[i];
        a[8 + i] = hi[i];
    }
    return a;
}

// modes: 0 = plain (+f16 copy to outh)   1 = sigmoid
//        2 = sigmoid(.)*aux1 -> outf and f16 to outh   3 = final: z*h+(1-z)*tanh(.)
__global__ void k_gemm64(const h16* __restrict__ A, int lda, int K,
                         const h16* __restrict__ B, const float* __restrict__ bias,
                         int mode, const float* __restrict__ aux1, const float* __restrict__ aux2,
                         float* __restrict__ outf,
                         h16* __restrict__ outh, int outh_stride, int outh_off,
                         int ntiles) {
    __shared__ __attribute__((aligned(32))) h16 sB[192 * 64];  // fragment-swizzled
    int tid = threadIdx.x;
    for (int i = tid; i < K * 64; i += blockDim.x) {
        int k = i >> 6, n = i & 63;
        int grp  = (k >> 5) * 4 + (n >> 4);            // (k-block, col-group)
        int lane = (n & 15) | (((k >> 4) & 1) << 4);
        int slot = k & 15;
        sB[(grp * 32 + lane) * 16 + slot] = B[i];
    }
    __syncthreads();

    int wave = tid >> 5, lane = tid & 31;
    int tile = blockIdx.x * (blockDim.x >> 5) + wave;
    if (tile >= ntiles) return;
    long row0 = (long)tile * 16;

    v8f acc[4] = {};
    for (int kb = 0; kb < K; kb += 32) {
        v16h a = load_a_frag(A, lda, row0, kb, lane);
        int kblk4 = (kb >> 5) * 4;
#pragma unroll
        for (int g = 0; g < 4; ++g) {
            v16h b = *(const v16h*)&sB[((kblk4 + g) * 32 + lane) * 16];
            acc[g] = __builtin_amdgcn_wmma_f32_16x16x32_f16(
                false, a, false, b, (short)0, acc[g], false, false);
        }
    }

    int colb = lane & 15;
    int rowadd = (lane >> 4) * 8;
#pragma unroll
    for (int g = 0; g < 4; ++g) {
#pragma unroll
        for (int r = 0; r < 8; ++r) {
            long row = row0 + r + rowadd;
            int  col = g * 16 + colb;
            float v = acc[g][r] + bias[col];
            long idx = row * 64 + col;
            if (mode == 0) {
                outf[idx] = v;
                outh[row * (long)outh_stride + outh_off + col] = (h16)v;
            } else if (mode == 1) {
                outf[idx] = 1.0f / (1.0f + expf(-v));
            } else if (mode == 2) {
                float s = 1.0f / (1.0f + expf(-v));
                float val = s * aux1[idx];
                outf[idx] = val;
                outh[row * (long)outh_stride + outh_off + col] = (h16)val;
            } else {
                float ht = tanhf(v);
                float z = aux1[idx];
                outf[idx] = z * aux2[idx] + (1.0f - z) * ht;
            }
        }
    }
}

// ---------------------------------------------------------------- launcher

static inline char* carve(char*& p, size_t bytes) {
    char* r = p;
    p += (bytes + 255) & ~(size_t)255;
    return r;
}

extern "C" void kernel_launch(void* const* d_in, const int* in_sizes, int n_in,
                              void* d_out, int out_size, void* d_ws, size_t ws_size,
                              hipStream_t stream) {
    const float*     x   = (const float*)d_in[0];
    const long long* ei  = (const long long*)d_in[1];
    const float*     h   = (const float*)d_in[2];
    const float*     W1  = (const float*)d_in[3];
    const float*     b1  = (const float*)d_in[4];
    const float*     W2  = (const float*)d_in[5];
    const float*     b2  = (const float*)d_in[6];
    const float*     Wz  = (const float*)d_in[7];
    const float*     bz  = (const float*)d_in[8];
    const float*     Wr  = (const float*)d_in[9];
    const float*     br  = (const float*)d_in[10];
    const float*     Wc  = (const float*)d_in[11];
    const float*     bc  = (const float*)d_in[12];
    const long long* src = ei;
    const long long* dst = ei + NE;

    char* p = (char*)d_ws;
    float* deg  = (float*)carve(p, (size_t)NN * 4);           // -> dinv in place
    float* nrm  = (float*)carve(p, (size_t)NE * 4);
    float* tx1  = (float*)carve(p, (size_t)NN * 2 * 4);
    float* tx2  = (float*)carve(p, (size_t)NN * 2 * 4);
    float* th1  = (float*)carve(p, (size_t)NN * 64 * 4);      // reused as z buffer
    float* th2  = (float*)carve(p, (size_t)NN * 64 * 4);      // reused as r*hc scratch
    float* hc   = (float*)carve(p, (size_t)NN * 64 * 4);
    h16*   Ah   = (h16*)carve(p, (size_t)NN * 192 * 2);
    h16*   comb = (h16*)carve(p, (size_t)NN * 128 * 2);
    h16*   cmbc = (h16*)carve(p, (size_t)NN * 128 * 2);
    h16*   Weff = (h16*)carve(p, (size_t)192 * 64 * 2);
    h16*   Wzh  = (h16*)carve(p, (size_t)128 * 64 * 2);
    h16*   Wrh  = (h16*)carve(p, (size_t)128 * 64 * 2);
    h16*   Wch  = (h16*)carve(p, (size_t)128 * 64 * 2);
    float* zbuf = th1;
    float* rscr = th2;

    const int T = 256;
    // zero accumulation buffers
    k_zero<<<2048, T, 0, stream>>>(deg, (size_t)NN);
    k_zero<<<2048, T, 0, stream>>>(tx1, (size_t)NN * 2);
    k_zero<<<2048, T, 0, stream>>>(tx2, (size_t)NN * 2);
    k_zero<<<4096, T, 0, stream>>>(th1, (size_t)NN * 64);
    k_zero<<<4096, T, 0, stream>>>(th2, (size_t)NN * 64);

    // weight conversions (independent)
    k_prep_weff<<<(64 * 64 + T - 1) / T, T, 0, stream>>>(W2, Weff);
    k_cvt_h<<<(128 * 64 + T - 1) / T, T, 0, stream>>>(Wz, Wzh, 128 * 64);
    k_cvt_h<<<(128 * 64 + T - 1) / T, T, 0, stream>>>(Wr, Wrh, 128 * 64);
    k_cvt_h<<<(128 * 64 + T - 1) / T, T, 0, stream>>>(Wc, Wch, 128 * 64);

    // normalization
    k_degree<<<(NE + T - 1) / T, T, 0, stream>>>(src, deg, NE);
    k_dinv<<<(NN + T - 1) / T, T, 0, stream>>>(deg, NN);
    k_edge_norm<<<(NE + T - 1) / T, T, 0, stream>>>(src, dst, deg, nrm, NE);

    // input ChebConv (2-wide)
    k_prop2<<<(NE + T - 1) / T, T, 0, stream>>>(src, dst, nrm, x, tx1, NE);
    k_prop2<<<(NE + T - 1) / T, T, 0, stream>>>(src, dst, nrm, tx1, tx2, NE);
    k_input_conv<<<(NN * 64 + T - 1) / T, T, 0, stream>>>(x, tx1, tx2, W1, b1, comb, cmbc, NN);

    // hidden ChebConv propagation (64-wide, HBM/atomic bound)
    long tot = (long)NE * 64;
    k_prop64<<<(int)((tot + T - 1) / T), T, 0, stream>>>(src, dst, nrm, h, th1, tot);
    k_prop64<<<(int)((tot + T - 1) / T), T, 0, stream>>>(src, dst, nrm, th1, th2, tot);
    k_pack_Ah<<<(NN * 64 + T - 1) / T, T, 0, stream>>>(h, th1, th2, Ah, NN);

    const int ntiles = NN / 16;                 // 6250
    const int waves_per_blk = 8;
    const int ggrid = (ntiles + waves_per_blk - 1) / waves_per_blk;

    // hidden_conv = Ah @ Weff + b2 ; f16 copy into combined[:,64:128]
    k_gemm64<<<ggrid, 256, 0, stream>>>(Ah, 192, 192, Weff, b2, 0,
                                        nullptr, nullptr, hc, comb, 128, 64, ntiles);
    // z = sigmoid(combined @ Wz + bz)
    k_gemm64<<<ggrid, 256, 0, stream>>>(comb, 128, 128, Wzh, bz, 1,
                                        nullptr, nullptr, zbuf, nullptr, 0, 0, ntiles);
    // combC[:,64:] = sigmoid(combined @ Wr + br) * hc
    k_gemm64<<<ggrid, 256, 0, stream>>>(comb, 128, 128, Wrh, br, 2,
                                        hc, nullptr, rscr, cmbc, 128, 64, ntiles);
    // out = z*h + (1-z)*tanh(combC @ Wc + bc)
    k_gemm64<<<ggrid, 256, 0, stream>>>(cmbc, 128, 128, Wch, bc, 3,
                                        zbuf, h, (float*)d_out, nullptr, 0, 0, ntiles);

    (void)in_sizes; (void)n_in; (void)out_size; (void)ws_size;
}
